// My_GRU_40913858462312
// MI455X (gfx1250) — matrix operand order repaired and checked by reading, not compiled, not run on hardware
//
#include <hip/hip_runtime.h>
#include <hip/hip_bf16.h>
#include <math.h>

// ---------------------------------------------------------------------------
// GRU recurrence, persistent kernel, bf16 WMMA (v_wmma_f32_16x16x32_bf16).
// B=256, T=512, I=512, H=512, O=1.
//  - 16 workgroups x 512 threads (16 wave32), each WG owns 16 batch rows.
//  - h kept in fp32 registers with the same (m,n)<->(lane,vgpr) mapping as the
//    WMMA C tiles, so gate/blend math never leaves registers.
//  - Weights pre-packed (one-time kernels) into per-lane B-fragment order.
//  - k-loops rolled (#pragma unroll 1): one LDS A-frag feeds 4 WMMAs, B frags
//    stream from L2-resident packed weights (no hoisting/spills).
//  - Next-step x tile prefetched (global_prefetch) to hide HBM latency behind
//    the matmul phases.
// ---------------------------------------------------------------------------

typedef __attribute__((ext_vector_type(16))) __bf16 v16bf;
typedef __attribute__((ext_vector_type(8)))  float  v8f;

#define Bsz 256
#define Tt  512
#define Ii  512
#define Hh  512

__device__ __forceinline__ unsigned short f2bf(float x) {
    unsigned int u = __float_as_uint(x);
    u += 0x7FFFu + ((u >> 16) & 1u);          // round-to-nearest-even
    return (unsigned short)(u >> 16);
}

union AB16 { uint4 q[2]; v16bf v; };

// A-fragment (16x32 bf16) from row-major LDS [16][512] bf16.
// 16-bit A layout: lanes 0-15 row M=lane: K = {k0..k0+7, k0+16..k0+23}, k0=0;
// lanes 16-31 same rows, k0=8.
__device__ __forceinline__ v16bf lds_a_frag(const unsigned short* base, int lane, int kt) {
    int row = lane & 15;
    int k0  = (kt << 5) + ((lane >> 4) << 3);
    const unsigned short* p = base + row * Hh + k0;
    AB16 f;
    f.q[0] = *(const uint4*)(p);        // K = k0 .. k0+7
    f.q[1] = *(const uint4*)(p + 16);   // K = k0+16 .. k0+23
    return f.v;
}

// B-fragment (32x16 bf16) at a pre-packed tile pointer (lane already folded in):
// tile = 32 lanes x 16 bf16 contiguous.
__device__ __forceinline__ v16bf b_frag_at(const unsigned short* __restrict__ p) {
    AB16 f;
    f.q[0] = *(const uint4*)(p);
    f.q[1] = *(const uint4*)(p + 8);
    return f.v;
}

__device__ __forceinline__ v8f wmma_bf16(v16bf a, v16bf b, v8f c) {
    return __builtin_amdgcn_wmma_f32_16x16x32_bf16(
        /*neg_a=*/false, a, /*neg_b=*/false, b,
        /*c_mod=*/(short)0, c, /*reuse_a=*/false, /*reuse_b=*/false);
}

__device__ __forceinline__ float sigmoidf_(float x) {
    return 1.0f / (1.0f + __expf(-x));
}
__device__ __forceinline__ float tanhf_(float x) {
    x = fminf(fmaxf(x, -15.0f), 15.0f);
    float e = __expf(2.0f * x);
    return (e - 1.0f) / (e + 1.0f);
}

#define V8ZERO {0.f,0.f,0.f,0.f,0.f,0.f,0.f,0.f}
#define TILE_ELEMS 512   // one packed B tile = 32 lanes * 16 bf16

// ------------------------ weight pack kernel --------------------------------
// dst element id -> (tile = nt*ktiles + kt, lane, j); src = W[k][n] row-major.
__global__ void pack_w_kernel(const float* __restrict__ W, unsigned short* __restrict__ dst,
                              int K, int N) {
    int ktiles = K >> 5;
    long id = (long)blockIdx.x * blockDim.x + threadIdx.x;
    long total = (long)K * (long)N;
    if (id >= total) return;
    int  j    = (int)(id & 15);
    int  lane = (int)((id >> 4) & 31);
    long tile = id >> 9;
    int  kt   = (int)(tile % ktiles);
    int  nt   = (int)(tile / ktiles);
    int  k    = kt * 32 + ((lane >> 4) << 3) + (j & 7) + ((j >> 3) << 4);
    int  n    = nt * 16 + (lane & 15);
    dst[id] = f2bf(W[(long)k * N + n]);
}

// ------------------------ persistent GRU kernel -----------------------------
__global__ __launch_bounds__(512)
void gru_persistent_kernel(const float* __restrict__ x,
                           const float* __restrict__ bg,
                           const float* __restrict__ bh,
                           const float* __restrict__ bo1,
                           const float* __restrict__ Wo2,
                           const float* __restrict__ bo2,
                           const unsigned short* __restrict__ WgP,
                           const unsigned short* __restrict__ WhP,
                           const unsigned short* __restrict__ Wo1P,
                           float* __restrict__ out) {
    __shared__ unsigned short xbf [16 * Hh];   // x_t tile, bf16 row-major
    __shared__ unsigned short hbf [16 * Hh];   // h tile, bf16 row-major
    __shared__ unsigned short rhbf[16 * Hh];   // reset*h (phases 2-3) / tmp fp32 (phases 4-5)
    __shared__ float          red [16 * 16];

    float* tmp = (float*)rhbf;                 // 16 x 256 fp32 == 16KB, same storage

    const int tid   = threadIdx.x;
    const int lane  = tid & 31;
    const int w     = tid >> 5;                // wave id 0..15
    const int br0   = blockIdx.x * 16;         // batch-tile origin
    const int mrow0 = (lane >> 4) << 3;        // 0 or 8
    const int ncol  = lane & 15;

    // Per-wave packed-weight base pointers (lane folded in).
    // Wg: 64 n-tiles x 32 k-tiles;  Wh: 32 n-tiles x 32 k-tiles; Wo1: 16 x 16.
    const unsigned short* gB0 = WgP  + (size_t)(2 * w + 0)      * 32 * TILE_ELEMS + lane * 16;
    const unsigned short* gB1 = WgP  + (size_t)(2 * w + 1)      * 32 * TILE_ELEMS + lane * 16;
    const unsigned short* gB2 = WgP  + (size_t)(32 + 2 * w + 0) * 32 * TILE_ELEMS + lane * 16;
    const unsigned short* gB3 = WgP  + (size_t)(32 + 2 * w + 1) * 32 * TILE_ELEMS + lane * 16;
    const unsigned short* hB0 = WhP  + (size_t)(2 * w + 0)      * 32 * TILE_ELEMS + lane * 16;
    const unsigned short* hB1 = WhP  + (size_t)(2 * w + 1)      * 32 * TILE_ELEMS + lane * 16;
    const unsigned short* oB0 = Wo1P + (size_t)w                * 16 * TILE_ELEMS + lane * 16;

    // fp32 hidden state in registers: hreg[j*8+r] <-> h[mrow0+r][32w+16j+ncol]
    float hreg[16];
#pragma unroll
    for (int i = 0; i < 16; ++i) hreg[i] = 0.0f;

    for (int i = tid; i < 16 * Hh; i += 512) hbf[i] = 0;
    __syncthreads();

    for (int t = 0; t < Tt; ++t) {
        // ---- phase 1: stage x_t tile (16x512 fp32 -> bf16 LDS) ----
        for (int c = tid; c < 2048; c += 512) {       // 2048 float4 chunks
            int m  = c >> 7;
            int i4 = c & 127;
            const float4 v = *(const float4*)(x + ((size_t)(br0 + m) * Tt + t) * Ii + (size_t)i4 * 4);
            ushort4 sv = make_ushort4(f2bf(v.x), f2bf(v.y), f2bf(v.z), f2bf(v.w));
            *(ushort4*)(xbf + m * Hh + i4 * 4) = sv;
        }
        __syncthreads();

        // Prefetch next step's x tile so HBM latency hides behind the matmuls.
        if (t + 1 < Tt) {
            for (int c = tid; c < 2048; c += 512) {
                int m  = c >> 7;
                int i4 = c & 127;
                __builtin_prefetch(x + ((size_t)(br0 + m) * Tt + (t + 1)) * Ii + (size_t)i4 * 4,
                                   0, 3);
            }
        }

        // ---- phase 2: gates = [x_t, h] @ Wg + bg ; 4 n-tiles in one k-loop ----
        v8f acc0 = V8ZERO, acc1 = V8ZERO, acc2 = V8ZERO, acc3 = V8ZERO;
#pragma unroll 1
        for (int kt = 0; kt < 16; ++kt) {              // K = x half
            v16bf a = lds_a_frag(xbf, lane, kt);
            size_t off = (size_t)kt * TILE_ELEMS;
            acc0 = wmma_bf16(a, b_frag_at(gB0 + off), acc0);
            acc1 = wmma_bf16(a, b_frag_at(gB1 + off), acc1);
            acc2 = wmma_bf16(a, b_frag_at(gB2 + off), acc2);
            acc3 = wmma_bf16(a, b_frag_at(gB3 + off), acc3);
        }
#pragma unroll 1
        for (int kt = 0; kt < 16; ++kt) {              // K = h half
            v16bf a = lds_a_frag(hbf, lane, kt);
            size_t off = (size_t)(16 + kt) * TILE_ELEMS;
            acc0 = wmma_bf16(a, b_frag_at(gB0 + off), acc0);
            acc1 = wmma_bf16(a, b_frag_at(gB1 + off), acc1);
            acc2 = wmma_bf16(a, b_frag_at(gB2 + off), acc2);
            acc3 = wmma_bf16(a, b_frag_at(gB3 + off), acc3);
        }

        float upd[16];
        {
            float bg0 = bg[(2 * w + 0) * 16 + ncol];
            float bg1 = bg[(2 * w + 1) * 16 + ncol];
#pragma unroll
            for (int r = 0; r < 8; ++r) {
                upd[r]     = sigmoidf_(acc0[r] + bg0);
                upd[8 + r] = sigmoidf_(acc1[r] + bg1);
            }
            int col0 = (2 * w + 0) * 16 + ncol;
            int col1 = (2 * w + 1) * 16 + ncol;
            float br0_ = bg[512 + col0];
            float br1_ = bg[512 + col1];
#pragma unroll
            for (int r = 0; r < 8; ++r) {
                float rs0 = sigmoidf_(acc2[r] + br0_);
                float rs1 = sigmoidf_(acc3[r] + br1_);
                rhbf[(mrow0 + r) * Hh + col0] = f2bf(rs0 * hreg[r]);
                rhbf[(mrow0 + r) * Hh + col1] = f2bf(rs1 * hreg[8 + r]);
            }
        }
        __syncthreads();

        // ---- phase 3: cand = tanh([x_t, r*h] @ Wh + bh); blend h in registers ----
        v8f cc0 = V8ZERO, cc1 = V8ZERO;
#pragma unroll 1
        for (int kt = 0; kt < 16; ++kt) {              // K = x half
            v16bf a = lds_a_frag(xbf, lane, kt);
            size_t off = (size_t)kt * TILE_ELEMS;
            cc0 = wmma_bf16(a, b_frag_at(hB0 + off), cc0);
            cc1 = wmma_bf16(a, b_frag_at(hB1 + off), cc1);
        }
#pragma unroll 1
        for (int kt = 0; kt < 16; ++kt) {              // K = reset*h half
            v16bf a = lds_a_frag(rhbf, lane, kt);
            size_t off = (size_t)(16 + kt) * TILE_ELEMS;
            cc0 = wmma_bf16(a, b_frag_at(hB0 + off), cc0);
            cc1 = wmma_bf16(a, b_frag_at(hB1 + off), cc1);
        }
        __syncthreads();   // done reading rhbf/xbf/hbf for this step's matmuls
        {
            int col0 = (2 * w + 0) * 16 + ncol;
            int col1 = (2 * w + 1) * 16 + ncol;
            float bh0 = bh[col0];
            float bh1 = bh[col1];
#pragma unroll
            for (int r = 0; r < 8; ++r) {
                float c0 = tanhf_(cc0[r] + bh0);
                float c1 = tanhf_(cc1[r] + bh1);
                float u0 = upd[r], u1 = upd[8 + r];
                float h0 = (1.0f - u0) * hreg[r]     + u0 * c0;
                float h1 = (1.0f - u1) * hreg[8 + r] + u1 * c1;
                hreg[r]     = h0;
                hreg[8 + r] = h1;
                hbf[(mrow0 + r) * Hh + col0] = f2bf(h0);  // h_t for out matmul + next step
                hbf[(mrow0 + r) * Hh + col1] = f2bf(h1);
            }
        }
        __syncthreads();

        // ---- phase 4: tmp = relu(h_t @ Wo1 + bo1), 16x256 ----
        {
            v8f acc = V8ZERO;
#pragma unroll 1
            for (int kt = 0; kt < 16; ++kt)
                acc = wmma_bf16(lds_a_frag(hbf, lane, kt),
                                b_frag_at(oB0 + (size_t)kt * TILE_ELEMS), acc);
            int col = w * 16 + ncol;
            float b1 = bo1[col];
#pragma unroll
            for (int r = 0; r < 8; ++r)
                tmp[(mrow0 + r) * 256 + col] = fmaxf(acc[r] + b1, 0.0f);
        }
        __syncthreads();

        // ---- phase 5: y = tmp @ Wo2 + bo2 (16x256 . 256) ----
        if (tid < 256) {
            int m = tid >> 4, seg = tid & 15;
            float s = 0.0f;
#pragma unroll
            for (int e = 0; e < 16; ++e)
                s += tmp[m * 256 + seg * 16 + e] * Wo2[seg * 16 + e];
            red[m * 16 + seg] = s;
        }
        __syncthreads();
        if (tid < 16) {
            float s = 0.0f;
#pragma unroll
            for (int k = 0; k < 16; ++k) s += red[tid * 16 + k];
            out[(size_t)(br0 + tid) * Tt + t] = s + bo2[0];
        }
        __syncthreads();
    }

    // ---- h_final (fp32) -> d_out + B*T ----
    float* hout = out + (size_t)Bsz * Tt;
#pragma unroll
    for (int j = 0; j < 2; ++j) {
#pragma unroll
        for (int r = 0; r < 8; ++r) {
            int m   = mrow0 + r;
            int col = (2 * w + j) * 16 + ncol;
            hout[(size_t)(br0 + m) * Hh + col] = hreg[j * 8 + r];
        }
    }
}

// ---------------------------------------------------------------------------
extern "C" void kernel_launch(void* const* d_in, const int* in_sizes, int n_in,
                              void* d_out, int out_size, void* d_ws, size_t ws_size,
                              hipStream_t stream) {
    (void)in_sizes; (void)n_in; (void)out_size; (void)ws_size;

    const float* x   = (const float*)d_in[0];
    const float* Wg  = (const float*)d_in[1];
    const float* bg  = (const float*)d_in[2];
    const float* Wh  = (const float*)d_in[3];
    const float* bh  = (const float*)d_in[4];
    const float* Wo1 = (const float*)d_in[5];
    const float* bo1 = (const float*)d_in[6];
    const float* Wo2 = (const float*)d_in[7];
    const float* bo2 = (const float*)d_in[8];
    float* out = (float*)d_out;

    // bf16 packed weights in workspace: 2MB + 1MB + 0.25MB
    unsigned short* WgP  = (unsigned short*)d_ws;
    unsigned short* WhP  = WgP + (size_t)1024 * 1024;
    unsigned short* Wo1P = WhP + (size_t)1024 * 512;

    pack_w_kernel<<<(1024 * 1024 + 255) / 256, 256, 0, stream>>>(Wg,  WgP,  1024, 1024);
    pack_w_kernel<<<(1024 * 512  + 255) / 256, 256, 0, stream>>>(Wh,  WhP,  1024, 512);
    pack_w_kernel<<<( 512 * 256  + 255) / 256, 256, 0, stream>>>(Wo1, Wo1P,  512, 256);

    gru_persistent_kernel<<<16, 512, 0, stream>>>(x, bg, bh, bo1, Wo2, bo2,
                                                  WgP, WhP, Wo1P, out);
}
